// BigBirdAttention_58969900974411
// MI455X (gfx1250) — matrix operand order repaired and checked by reading, not compiled, not run on hardware
//
#include <hip/hip_runtime.h>

// ---------------------------------------------------------------------------
// BigBird block-sparse attention for gfx1250 (MI455X), bf16 WMMA pipeline.
//   B=2, S=8192, HID=768, H=12, D=64, BS=64, NB=128, R=3, G=8 gathered blocks
// ---------------------------------------------------------------------------

typedef __attribute__((ext_vector_type(16))) __bf16 v16bf;
typedef __attribute__((ext_vector_type(8)))  float  v8f;
typedef int v4i __attribute__((vector_size(16)));
typedef unsigned short u16;
typedef unsigned int   u32;

#define BB    2
#define SEQ   8192
#define HID   768
#define NH    12
#define DH    64
#define BS    64
#define NBLK  128
#define RR    3

// ---- CDNA5 async global->LDS path (guarded; fallback = sync staging) ------
#if defined(__has_builtin)
#if __has_builtin(__builtin_amdgcn_global_load_async_to_lds_b128)
#define ATTN_ASYNC 1
#endif
#endif

#if defined(ATTN_ASYNC)
#if __has_builtin(__builtin_amdgcn_s_wait_asynccnt)
#define ASYNC_WAIT(n) __builtin_amdgcn_s_wait_asynccnt(n)
#else
#define ASYNC_WAIT(n) asm volatile("s_wait_asynccnt %0" ::"i"(n) : "memory")
#endif
#else
#define ASYNC_WAIT(n) ((void)0)
#endif

__device__ __forceinline__ void cp_async16(const u16* g, u16* l) {
#if defined(ATTN_ASYNC)
  __builtin_amdgcn_global_load_async_to_lds_b128(
      (__attribute__((address_space(1))) v4i*)(v4i*)const_cast<u16*>(g),
      (__attribute__((address_space(3))) v4i*)(v4i*)l, 0, 0);
#else
  *(uint4*)l = *(const uint4*)g;
#endif
}

// f32 -> bf16 round-to-nearest-even
__device__ __forceinline__ u16 f2bf(float f) {
  union { float f; u32 u; } c; c.f = f;
  u32 r = c.u + 0x7fffu + ((c.u >> 16) & 1u);
  return (u16)(r >> 16);
}

union Frag { v16bf v; uint4 q[2]; };

// ---------------------------------------------------------------------------
// Kernel 1: hidden f32 -> bf16 (element-preserving), 2 elems/thread
// ---------------------------------------------------------------------------
__global__ __launch_bounds__(256) void cvt_x(const float* __restrict__ X,
                                             u16* __restrict__ Xb, int n2) {
  int i = blockIdx.x * 256 + threadIdx.x;
  if (i < n2) {
    float2 f = ((const float2*)X)[i];
    u32 p = (u32)f2bf(f.x) | ((u32)f2bf(f.y) << 16);
    ((u32*)Xb)[i] = p;
  }
}

// ---------------------------------------------------------------------------
// Kernel 2: Wq/Wk/Wv f32 [k][n] -> bf16 transposed WT [mat][n][k]
// ---------------------------------------------------------------------------
__global__ __launch_bounds__(256) void cvt_wt(const float* __restrict__ Wq,
                                              const float* __restrict__ Wk,
                                              const float* __restrict__ Wv,
                                              u16* __restrict__ WT) {
  int i = blockIdx.x * 256 + threadIdx.x;
  if (i >= 3 * HID * HID) return;
  int mat = i / (HID * HID);
  int rem = i % (HID * HID);
  int nn = rem / HID, kk = rem % HID;
  const float* W = (mat == 0) ? Wq : (mat == 1) ? Wk : Wv;
  WT[i] = f2bf(W[kk * HID + nn]);
}

// ---------------------------------------------------------------------------
// Kernel 3: QKV projection GEMM.  Y = Xb @ W  (bf16 in, f32 acc, bf16 out)
//   one wave = 16 rows x 64 cols tile; K-loop 768 in steps of 32,
//   software-pipelined with ping-pong register fragments so loads for k+1
//   overlap WMMAs of k.
//   Q,K stored [b,h,s,d]; V stored transposed [b,h,d,s]; 1/sqrt(D) in Q.
// ---------------------------------------------------------------------------
__global__ __launch_bounds__(256) void qkv_gemm(const u16* __restrict__ Xb,
                                                const u16* __restrict__ WT,
                                                u16* __restrict__ Qo,
                                                u16* __restrict__ Ko,
                                                u16* __restrict__ VTo) {
  const int unit = blockIdx.x * 8 + (threadIdx.x >> 5);     // 36864 wave units
  const int lane = threadIdx.x & 31;
  const int m15 = lane & 15, khalf = lane >> 4;

  const int mat = unit / (1024 * 12);           // 0=Q 1=K 2=V
  const int rem = unit % (1024 * 12);
  const int mt  = rem / 12;                     // 0..1023 (16-row tiles)
  const int nt  = rem % 12;                     // 0..11 (64-col tiles)
  const int m0  = mt * 16;
  const int mrow = m0 + m15;

  const u16* W = WT + (size_t)mat * HID * HID;
  const u16* arow = Xb + (size_t)mrow * HID + khalf * 16;

  Frag a[2];
  Frag b[2][4];
  // prologue: fragments for kb = 0
  a[0].q[0] = *(const uint4*)(arow);
  a[0].q[1] = *(const uint4*)(arow + 8);
#pragma unroll
  for (int dt = 0; dt < 4; ++dt) {
    const u16* bp = W + (size_t)(nt * 64 + dt * 16 + m15) * HID + khalf * 16;
    b[0][dt].q[0] = ((const uint4*)bp)[0];
    b[0][dt].q[1] = ((const uint4*)bp)[1];
  }

  v8f acc[4] = {};
#pragma unroll
  for (int kb = 0; kb < 24; ++kb) {
    const int cur = kb & 1, nxt = cur ^ 1;
    if (kb + 1 < 24) {                          // prefetch next K-slice
      const int kbase = (kb + 1) * 32;
      a[nxt].q[0] = *(const uint4*)(arow + kbase);
      a[nxt].q[1] = *(const uint4*)(arow + kbase + 8);
#pragma unroll
      for (int dt = 0; dt < 4; ++dt) {
        const u16* bp = W + (size_t)(nt * 64 + dt * 16 + m15) * HID + kbase + khalf * 16;
        b[nxt][dt].q[0] = ((const uint4*)bp)[0];
        b[nxt][dt].q[1] = ((const uint4*)bp)[1];
      }
    }
#pragma unroll
    for (int dt = 0; dt < 4; ++dt) {
      acc[dt] = __builtin_amdgcn_wmma_f32_16x16x32_bf16(
          false, a[cur].v, false, b[cur][dt].v, (short)0, acc[dt], false, false);
    }
  }

  const float qscale = 0.125f;   // 1/sqrt(64), folded into Q only
#pragma unroll
  for (int dt = 0; dt < 4; ++dt) {
#pragma unroll
    for (int r = 0; r < 8; ++r) {
      const int row = m0 + r + 8 * khalf;       // global row in [0, B*S)
      const int gb = row >> 13;                 // /8192
      const int s  = row & (SEQ - 1);
      const int c  = nt * 64 + dt * 16 + m15;   // output col in [0,768)
      const int h  = c >> 6, d = c & 63;
      if (mat == 0) {
        Qo[(((size_t)(gb * NH + h) * SEQ) + s) * DH + d] = f2bf(acc[dt][r] * qscale);
      } else if (mat == 1) {
        Ko[(((size_t)(gb * NH + h) * SEQ) + s) * DH + d] = f2bf(acc[dt][r]);
      } else {
        VTo[(((size_t)(gb * NH + h) * DH) + d) * SEQ + s] = f2bf(acc[dt][r]);
      }
    }
  }
}

// ---------------------------------------------------------------------------
// Kernel 4: flash-style block-sparse attention.
//   1 workgroup per (b,h,qblock); 4 waves, each owns 16 query rows.
//   K/V chunks (32 keys) staged cooperatively into LDS with async
//   global->LDS copies, double-buffered one step ahead (ASYNCcnt).
//   Interior blocks: 8 gathered key blocks; blocks 0 / NB-1: dense.
//   Dense (16x longer) workgroups are mapped to the lowest blockIdx values
//   so they launch first and their tail hides under the short interior ones.
// ---------------------------------------------------------------------------
__global__ __launch_bounds__(128) void bigbird_attn(const u16* __restrict__ Q,
                                                    const u16* __restrict__ K,
                                                    const u16* __restrict__ VT,
                                                    const int* __restrict__ rb,
                                                    float* __restrict__ out) {
  __shared__ __align__(16) u16 kbuf[2][32 * DH];   // 2 x 4KB  [key][d]
  __shared__ __align__(16) u16 vbuf[2][DH * 32];   // 2 x 4KB  [d][key]
  __shared__ __align__(16) u16 pbuf[4][16 * 32];   // per-wave P tile

  // longest-job-first remap: first 48 blocks are the dense ones
  const int nDense = BB * NH * 2;                  // 48
  const int blk = blockIdx.x;
  int bh, nb;
  if (blk < nDense) {
    bh = blk >> 1;
    nb = (blk & 1) ? (NBLK - 1) : 0;
  } else {
    const int i = blk - nDense;
    bh = i / (NBLK - 2);
    nb = 1 + i % (NBLK - 2);
  }

  const int tid = threadIdx.x;
  const int wv = tid >> 5;
  const int lane = tid & 31;
  const int m15 = lane & 15, khalf = lane >> 4;

  const u16* Qh = Q  + (size_t)bh * SEQ * DH;
  const u16* Kh = K  + (size_t)bh * SEQ * DH;
  const u16* Vh = VT + (size_t)bh * DH * SEQ;

  // preload 2 Q A-fragments (d = 0..31, 32..63) for this wave's 16 rows
  const int qrow = nb * BS + wv * 16 + m15;
  Frag qa[2];
#pragma unroll
  for (int kk = 0; kk < 2; ++kk) {
    const u16* qp = Qh + (size_t)qrow * DH + kk * 32 + khalf * 16;
    qa[kk].q[0] = *(const uint4*)qp;
    qa[kk].q[1] = *(const uint4*)(qp + 8);
  }

  const bool dense = (nb == 0) || (nb == NBLK - 1);
  int list[8];
  if (!dense) {
    list[0] = 0;        list[1] = NBLK - 1;
    list[2] = nb - 1;   list[3] = nb;       list[4] = nb + 1;
    list[5] = rb[nb * RR + 0];
    list[6] = rb[nb * RR + 1];
    list[7] = rb[nb * RR + 2];
  }
  const int nsteps = dense ? (NBLK * 2) : 16;   // 32 keys per step

  // stage K/V chunk for `step` into buffer bsel: 4 async 16B ops per thread
  auto stage = [&](int step, int bsel) {
    const int kblk = dense ? (step >> 1) : list[step >> 1];
    const int ks = kblk * BS + (step & 1) * 32;
#pragma unroll
    for (int i = 0; i < 2; ++i) {
      const int idx = i * 128 + tid;            // 0..255
      const int key = idx >> 3, kp = idx & 7;   // 32 keys x 8 parts(16B)
      cp_async16(Kh + (size_t)(ks + key) * DH + kp * 8,
                 &kbuf[bsel][key * DH + kp * 8]);
      const int d = idx >> 2, vp = idx & 3;     // 64 d x 4 parts(16B)
      cp_async16(Vh + (size_t)d * SEQ + ks + vp * 8,
                 &vbuf[bsel][d * 32 + vp * 8]);
    }
  };

  float mrow[8], lrow[8];
#pragma unroll
  for (int r = 0; r < 8; ++r) { mrow[r] = -1e30f; lrow[r] = 0.0f; }
  v8f cacc[4] = {};

  stage(0, 0);                                   // prologue
  for (int step = 0; step < nsteps; ++step) {
    const int cur = step & 1;
    if (step + 1 < nsteps) {
      stage(step + 1, cur ^ 1);
      ASYNC_WAIT(4);                             // current chunk landed
    } else {
      ASYNC_WAIT(0);
    }
    __syncthreads();                             // all waves' copies visible

    // ---- scores: two 16-key tiles, each 2 WMMAs over D=64 (from LDS) ----
    v8f sc[2];
#pragma unroll
    for (int t = 0; t < 2; ++t) {
      v8f s = {};
#pragma unroll
      for (int kk = 0; kk < 2; ++kk) {
        Frag kf;
        const u16* kp = &kbuf[cur][(t * 16 + m15) * DH + kk * 32 + khalf * 16];
        kf.q[0] = ((const uint4*)kp)[0];
        kf.q[1] = ((const uint4*)kp)[1];
        s = __builtin_amdgcn_wmma_f32_16x16x32_bf16(
            false, qa[kk].v, false, kf.v, (short)0, s, false, false);
      }
      sc[t] = s;
    }

    // ---- online softmax update; emit P tile to LDS ----
#pragma unroll
    for (int r = 0; r < 8; ++r) {
      float v0 = sc[0][r], v1 = sc[1][r];      // row r+8*khalf, 2 cols/lane
      float t = fmaxf(v0, v1);
      t = fmaxf(t, __shfl_xor(t, 1, 32));
      t = fmaxf(t, __shfl_xor(t, 2, 32));
      t = fmaxf(t, __shfl_xor(t, 4, 32));
      t = fmaxf(t, __shfl_xor(t, 8, 32));      // row max over 16 lanes (half)
      const float mnew = fmaxf(mrow[r], t);
      const float rescale = __expf(mrow[r] - mnew);
      mrow[r] = mnew;
      const float p0 = __expf(v0 - mnew);
      const float p1 = __expf(v1 - mnew);
      float rs = p0 + p1;
      rs += __shfl_xor(rs, 1, 32);
      rs += __shfl_xor(rs, 2, 32);
      rs += __shfl_xor(rs, 4, 32);
      rs += __shfl_xor(rs, 8, 32);
      lrow[r] = lrow[r] * rescale + rs;
#pragma unroll
      for (int dt = 0; dt < 4; ++dt) cacc[dt][r] *= rescale;
      const int row = r + 8 * khalf;           // row within 16-row tile
      pbuf[wv][row * 32 + m15]      = f2bf(p0);
      pbuf[wv][row * 32 + 16 + m15] = f2bf(p1);
    }
    asm volatile("" ::: "memory");             // keep DS store->load ordered

    // ---- P C-layout -> A-layout via LDS (DS ops in-order per wave) ----
    Frag pa;
    const u16* pp = &pbuf[wv][m15 * 32 + khalf * 16];
    pa.q[0] = *(const uint4*)pp;
    pa.q[1] = *(const uint4*)(pp + 8);
    asm volatile("" ::: "memory");

    // ---- ctx += P @ V  (4 WMMAs, D tiles of 16, V from LDS) ----
#pragma unroll
    for (int dt = 0; dt < 4; ++dt) {
      Frag vf;
      const u16* vp = &vbuf[cur][(dt * 16 + m15) * 32 + khalf * 16];
      vf.q[0] = ((const uint4*)vp)[0];
      vf.q[1] = ((const uint4*)vp)[1];
      cacc[dt] = __builtin_amdgcn_wmma_f32_16x16x32_bf16(
          false, pa.v, false, vf.v, (short)0, cacc[dt], false, false);
    }
    __syncthreads();                           // protect buffer re-use
  }

  // ---- normalize and store f32 output [b, s, h*64+d] ----
  const int b = bh / NH, h = bh % NH;
#pragma unroll
  for (int dt = 0; dt < 4; ++dt) {
#pragma unroll
    for (int r = 0; r < 8; ++r) {
      const int s = nb * BS + wv * 16 + r + 8 * khalf;
      const int c = h * DH + dt * 16 + m15;
      out[((size_t)b * SEQ + s) * HID + c] = cacc[dt][r] / lrow[r];
    }
  }
}

// ---------------------------------------------------------------------------
// Launch
// ---------------------------------------------------------------------------
extern "C" void kernel_launch(void* const* d_in, const int* in_sizes, int n_in,
                              void* d_out, int out_size, void* d_ws, size_t ws_size,
                              hipStream_t stream) {
  const float* hidden = (const float*)d_in[0];
  const float* Wq = (const float*)d_in[1];
  const float* Wk = (const float*)d_in[2];
  const float* Wv = (const float*)d_in[3];
  const int*   rb = (const int*)d_in[4];
  float* out = (float*)d_out;

  char* ws = (char*)d_ws;
  const size_t XBF_OFF = 0;                                  // 25,165,824 B
  const size_t WT_OFF  = XBF_OFF + (size_t)BB * SEQ * HID * 2;
  const size_t Q_OFF   = WT_OFF + (size_t)3 * HID * HID * 2;
  const size_t K_OFF   = Q_OFF + (size_t)BB * SEQ * HID * 2;
  const size_t VT_OFF  = K_OFF + (size_t)BB * SEQ * HID * 2;

  u16* Xb  = (u16*)(ws + XBF_OFF);
  u16* WT  = (u16*)(ws + WT_OFF);
  u16* Qb  = (u16*)(ws + Q_OFF);
  u16* Kb  = (u16*)(ws + K_OFF);
  u16* VTb = (u16*)(ws + VT_OFF);

  const int n2 = BB * SEQ * HID / 2;                         // 6,291,456
  cvt_x<<<(n2 + 255) / 256, 256, 0, stream>>>(hidden, Xb, n2);

  const int nw = 3 * HID * HID;                              // 1,769,472
  cvt_wt<<<(nw + 255) / 256, 256, 0, stream>>>(Wq, Wk, Wv, WT);

  // 3 mats * 1024 row-tiles * 12 col-tiles = 36864 wave units / 8 per block
  qkv_gemm<<<4608, 256, 0, stream>>>(Xb, WT, Qb, Kb, VTb);

  // B*H*NB = 3072 workgroups (48 dense first), 4 waves each
  bigbird_attn<<<BB * NH * NBLK, 128, 0, stream>>>(Qb, Kb, VTb, rb, out);
}
